// MultiCrossAttention_45964740002236
// MI455X (gfx1250) — matrix operand order repaired and checked by reading, not compiled, not run on hardware
//
#include <hip/hip_runtime.h>
#include <hip/hip_bf16.h>
#include <stdint.h>

// MultiCrossAttention for MI455X (gfx1250, wave32, WMMA + TDM).
// Pipeline: [prep: w->bf16 transposed] ->
//           [kv-proj GEMM: TDM double-buffered B tile in LDS, 8 waves share] ->
//           [flash attention, WMMA QK^T + WMMA PV, LDS relayout] ->
//           [out-proj GEMM: same TDM staging].
// All matrix math on v_wmma_f32_16x16x32_bf16 (fp32 accumulate).

typedef __attribute__((ext_vector_type(16))) __bf16      v16bf;
typedef __attribute__((ext_vector_type(8)))  float       v8f;
typedef __attribute__((ext_vector_type(4)))  unsigned int u32x4;
typedef __attribute__((ext_vector_type(4)))  float        f32x4;
typedef __attribute__((ext_vector_type(8)))  int          i32x8;
typedef __attribute__((ext_vector_type(4)))  int          i32x4;

union Frag {
    v16bf v;
    u32x4 u4[2];
    unsigned int u[8];
};

__device__ __forceinline__ unsigned short f2bf(float f) {
    unsigned int u = __builtin_bit_cast(unsigned int, f);
    u += 0x7FFFu + ((u >> 16) & 1u);   // round-to-nearest-even
    return (unsigned short)(u >> 16);
}

__device__ __forceinline__ unsigned int pack2(float lo, float hi) {
    return (unsigned int)f2bf(lo) | ((unsigned int)f2bf(hi) << 16);
}

// B-fragment (K x 16, lane = col n, element = K; lanes 0-15: K 0..15,
// lanes 16-31: K 16..31): 16 contiguous bf16 = 32 bytes.
__device__ __forceinline__ Frag load_b_bf16(const unsigned short* p) {
    Frag f;
    f.u4[0] = *(const u32x4*)(p);
    f.u4[1] = *(const u32x4*)(p + 8);
    return f;
}

// A-fragment (16 x 32, lane = row) from fp32 with bf16 convert.
// e=0..15 -> K = hi*16+e (callers pass p0 = base+hi*16, p1 = p0+8).
__device__ __forceinline__ void load_a_f32(Frag& f, const float* p0, const float* p1) {
    f32x4 a = *(const f32x4*)(p0);
    f32x4 b = *(const f32x4*)(p0 + 4);
    f32x4 c = *(const f32x4*)(p1);
    f32x4 d = *(const f32x4*)(p1 + 4);
    f.u[0] = pack2(a[0], a[1]); f.u[1] = pack2(a[2], a[3]);
    f.u[2] = pack2(b[0], b[1]); f.u[3] = pack2(b[2], b[3]);
    f.u[4] = pack2(c[0], c[1]); f.u[5] = pack2(c[2], c[3]);
    f.u[6] = pack2(d[0], d[1]); f.u[7] = pack2(d[2], d[3]);
}

__device__ __forceinline__ void load_a_bf16(Frag& f, const unsigned short* p0,
                                            const unsigned short* p1) {
    f.u4[0] = *(const u32x4*)(p0);
    f.u4[1] = *(const u32x4*)(p1);
}

__device__ __forceinline__ v8f wmma_bf16(const Frag& a, const Frag& b, v8f c) {
    return __builtin_amdgcn_wmma_f32_16x16x32_bf16(
        false, a.v, false, b.v, (short)0, c, false, false);
}

// ---------------------------------------------------------------------------
// TDM: async 2-D tile load (global -> LDS).  D# built per CDNA5 ISA §8:
//  g0: count=1 | lds_addr | global_addr(57b) | type=2
//  g1: data_size=2B | tensor dims (set huge: tile never clipped) |
//      tile_dim0 (contiguous elems) | tile_dim1 (rows) | row stride (elems)
//  groups 2/3 zero (2-D tensor).
// ---------------------------------------------------------------------------
__device__ __forceinline__ void tdm_load_2d(unsigned int lds_off, const void* gptr,
                                            unsigned int tile_d0, unsigned int tile_d1,
                                            unsigned int stride0) {
    unsigned long long ga = (unsigned long long)gptr;
    u32x4 g0;
    g0[0] = 1u;                                    // count=1 (valid user D#)
    g0[1] = lds_off;                               // LDS byte address
    g0[2] = (unsigned int)ga;                      // global_addr[31:0]
    g0[3] = (unsigned int)(ga >> 32) | (2u << 30); // global_addr[56:32] | type=2
    i32x8 g1;
    g1[0] = (int)(1u << 16);                       // data_size = 1 -> 2 bytes
    g1[1] = 0;                                     // abar=0 | tensor_dim0 lo16 (2^30 -> 0)
    g1[2] = (int)0x4000u;                          // tensor_dim0 hi16 | tensor_dim1 lo16
    g1[3] = (int)(0x4000u | (tile_d0 << 16));      // tensor_dim1 hi16 | tile_dim0
    g1[4] = (int)tile_d1;                          // tile_dim1 | tile_dim2=0
    g1[5] = (int)stride0;                          // tensor_dim0_stride lo32
    g1[6] = 0;                                     // stride0 hi16 | stride1 lo16
    g1[7] = 0;
    i32x4 gz = {0, 0, 0, 0};
#if __clang_major__ >= 23
    i32x8 gz8 = {0, 0, 0, 0, 0, 0, 0, 0};
    __builtin_amdgcn_tensor_load_to_lds(g0, g1, gz, gz, gz8, 0);
#else
    __builtin_amdgcn_tensor_load_to_lds(g0, g1, gz, gz, 0);
#endif
}

#define BB   2
#define TT   2048
#define DD   1024
#define HH   16
#define HD   64

// ---------------------------------------------------------------------------
// Kernel 0: convert + transpose weights to bf16 column-major.
// ---------------------------------------------------------------------------
__global__ void __launch_bounds__(256) prep_kernel(
    const float* __restrict__ kv_w, const float* __restrict__ out_w,
    unsigned short* __restrict__ kvwT, unsigned short* __restrict__ owT)
{
    const int stride = gridDim.x * blockDim.x;
    int i = blockIdx.x * blockDim.x + threadIdx.x;
    for (int idx = i; idx < 2 * DD * DD; idx += stride) {
        int n = idx >> 10, k = idx & 1023;
        kvwT[idx] = f2bf(kv_w[(size_t)k * (2 * DD) + n]);
    }
    for (int idx = i; idx < DD * DD; idx += stride) {
        int n = idx >> 10, k = idx & 1023;
        owT[idx] = f2bf(out_w[(size_t)k * DD + n]);
    }
}

// ---------------------------------------------------------------------------
// Kernel 1: kv = x @ kv_w + kv_b -> Kh[b][h][t][hd], Vt[b][h][hd][t] (bf16).
// Block = 8 waves: 128(M) x 128(N) tile; wave w owns rows [tm+16w, tm+16w+16).
// B tile (128 n-rows x 32 k) is TDM-DMA'd into double-buffered LDS and shared
// by all 8 waves; next k-step DMA overlaps current WMMAs.
// ---------------------------------------------------------------------------
__global__ void __launch_bounds__(256) kv_proj_kernel(
    const float* __restrict__ x, const float* __restrict__ kv_b,
    const unsigned short* __restrict__ kvwT,
    unsigned short* __restrict__ Kh, unsigned short* __restrict__ Vt)
{
    __shared__ unsigned short Bst[2][128 * 32];    // 2 x 8KB

    const int lane = threadIdx.x & 31;
    const int wave = threadIdx.x >> 5;
    const int half = lane >> 4;
    const int l16  = lane & 15;
    const int tm   = ((blockIdx.x >> 4) << 7) + (wave << 4);   // 32 M-tiles of 128
    const int tn   = (blockIdx.x & 15) << 7;                   // 16 N-tiles of 128

    v8f acc[8] = {};
    const float* xr = x + (size_t)(tm + l16) * DD;
    const unsigned short* wbase = kvwT + (size_t)tn * DD;      // row n' stride DD

    if (wave == 0)
        tdm_load_2d((unsigned int)(size_t)&Bst[0][0], wbase, 32, 128, DD);

    for (int it = 0; it < 32; ++it) {
        const int kb  = it << 5;
        const int cur = it & 1;
        if (wave == 0)
            __builtin_amdgcn_s_wait_tensorcnt(0);   // buf[cur] resident
        __syncthreads();
        if (wave == 0 && it + 1 < 32)               // prefetch next k-step
            tdm_load_2d((unsigned int)(size_t)&Bst[1 - cur][0],
                        wbase + kb + 32, 32, 128, DD);

        Frag a;
        load_a_f32(a, xr + kb + half * 16, xr + kb + half * 16 + 8);
        const unsigned short* bs = &Bst[cur][0] + l16 * 32 + half * 16;
#pragma unroll
        for (int j = 0; j < 8; ++j) {
            Frag b;
            b.u4[0] = *(const u32x4*)(bs + j * 512);
            b.u4[1] = *(const u32x4*)(bs + j * 512 + 8);
            acc[j] = wmma_bf16(a, b, acc[j]);
        }
        __syncthreads();   // all reads of buf[cur] done before it is re-DMA'd
    }

#pragma unroll
    for (int j = 0; j < 8; ++j) {
        const int gn = tn + j * 16 + l16;
        const float bias = kv_b[gn];
#pragma unroll
        for (int v = 0; v < 8; ++v) {
            const int gm = tm + v + half * 8;          // global row = b*T + t
            const int bb = gm >> 11;
            const int t  = gm & (TT - 1);
            const unsigned short bv = f2bf(acc[j][v] + bias);
            if (gn < DD) {
                const int h = gn >> 6, hd = gn & 63;
                Kh[((size_t)(bb * HH + h) * TT + t) * HD + hd] = bv;
            } else {
                const int g = gn - DD;
                const int h = g >> 6, hd = g & 63;
                Vt[((size_t)(bb * HH + h) * HD + hd) * TT + t] = bv;
            }
        }
    }
}

// ---------------------------------------------------------------------------
// Kernel 2: flash attention.  One wave owns (b, h, 16 q-rows).
// ---------------------------------------------------------------------------
__global__ void __launch_bounds__(256) attn_kernel(
    const float* __restrict__ q,
    const unsigned short* __restrict__ Kh, const unsigned short* __restrict__ Vt,
    unsigned short* __restrict__ attn)
{
    __shared__ unsigned short Psh[8][16 * 32];     // per-wave P tile (bf16)

    const int lane = threadIdx.x & 31;
    const int wave = threadIdx.x >> 5;
    const int tile = blockIdx.x * 8 + wave;        // 4096 tiles
    const int bh   = tile >> 7;                    // b*H + h
    const int q0   = (tile & 127) << 4;
    const int b    = bh >> 4;
    const int h    = bh & 15;
    const int half = lane >> 4;
    const int l16  = lane & 15;

    const float* qr = q + ((size_t)b * TT + q0 + l16) * DD + h * HD;
    Frag A0, A1;
    load_a_f32(A0, qr + half * 16,      qr + half * 16 + 8);
    load_a_f32(A1, qr + 32 + half * 16, qr + 40 + half * 16);

    const unsigned short* Kb = Kh + (size_t)bh * TT * HD;   // [t][hd]
    const unsigned short* Vb = Vt + (size_t)bh * HD * TT;   // [hd][t]
    unsigned short* Pw = Psh[wave];

    v8f O[4] = {};
    float m[8], l[8];
#pragma unroll
    for (int v = 0; v < 8; ++v) { m[v] = -3.0e38f; l[v] = 0.0f; }

    for (int c0 = 0; c0 < TT; c0 += 32) {
        const unsigned short* kra = Kb + (size_t)(c0 + l16) * HD;
        const unsigned short* krb = Kb + (size_t)(c0 + 16 + l16) * HD;
        Frag Ba0 = load_b_bf16(kra + half * 16);
        Frag Ba1 = load_b_bf16(kra + 32 + half * 16);
        Frag Bb0 = load_b_bf16(krb + half * 16);
        Frag Bb1 = load_b_bf16(krb + 32 + half * 16);
        Frag Bv[4];
#pragma unroll
        for (int nt = 0; nt < 4; ++nt)
            Bv[nt] = load_b_bf16(Vb + (size_t)(nt * 16 + l16) * TT + c0 + half * 16);

        v8f Sa = {}; Sa = wmma_bf16(A0, Ba0, Sa); Sa = wmma_bf16(A1, Ba1, Sa);
        v8f Sb = {}; Sb = wmma_bf16(A0, Bb0, Sb); Sb = wmma_bf16(A1, Bb1, Sb);

#pragma unroll
        for (int v = 0; v < 8; ++v) {
            const float sa = Sa[v] * 0.125f;
            const float sb = Sb[v] * 0.125f;
            float t = fmaxf(sa, sb);
            t = fmaxf(t, __shfl_xor(t, 1, 32));
            t = fmaxf(t, __shfl_xor(t, 2, 32));
            t = fmaxf(t, __shfl_xor(t, 4, 32));
            t = fmaxf(t, __shfl_xor(t, 8, 32));
            const float mn    = fmaxf(m[v], t);
            const float alpha = __expf(m[v] - mn);
            m[v] = mn;
            const float pa = __expf(sa - mn);
            const float pb = __expf(sb - mn);
            float rs = pa + pb;
            rs += __shfl_xor(rs, 1, 32);
            rs += __shfl_xor(rs, 2, 32);
            rs += __shfl_xor(rs, 4, 32);
            rs += __shfl_xor(rs, 8, 32);
            l[v] = l[v] * alpha + rs;
            O[0][v] *= alpha; O[1][v] *= alpha; O[2][v] *= alpha; O[3][v] *= alpha;
            const int row = v + half * 8;
            Pw[row * 32 + l16]      = f2bf(pa);
            Pw[row * 32 + 16 + l16] = f2bf(pb);
        }

        Frag P;
        const unsigned short* pr = Pw + l16 * 32;
        load_a_bf16(P, pr + half * 16, pr + half * 16 + 8);

#pragma unroll
        for (int nt = 0; nt < 4; ++nt)
            O[nt] = wmma_bf16(P, Bv[nt], O[nt]);
    }

#pragma unroll
    for (int v = 0; v < 8; ++v) {
        const float inv = 1.0f / l[v];
        const size_t row = (size_t)b * TT + q0 + v + half * 8;
#pragma unroll
        for (int nt = 0; nt < 4; ++nt)
            attn[row * DD + h * HD + nt * 16 + l16] = f2bf(O[nt][v] * inv);
    }
}

// ---------------------------------------------------------------------------
// Kernel 3: out = attn @ out_w + out_b (fp32).  Same TDM staging as kernel 1.
// Block: 128(M) x 128(N).
// ---------------------------------------------------------------------------
__global__ void __launch_bounds__(256) out_proj_kernel(
    const unsigned short* __restrict__ attn, const unsigned short* __restrict__ owT,
    const float* __restrict__ out_b, float* __restrict__ out)
{
    __shared__ unsigned short Bst[2][128 * 32];

    const int lane = threadIdx.x & 31;
    const int wave = threadIdx.x >> 5;
    const int half = lane >> 4;
    const int l16  = lane & 15;
    const int tm   = ((blockIdx.x >> 3) << 7) + (wave << 4);   // 32 M-tiles
    const int tn   = (blockIdx.x & 7) << 7;                    // 8 N-tiles

    v8f acc[8] = {};
    const unsigned short* ar = attn + (size_t)(tm + l16) * DD;
    const unsigned short* wbase = owT + (size_t)tn * DD;

    if (wave == 0)
        tdm_load_2d((unsigned int)(size_t)&Bst[0][0], wbase, 32, 128, DD);

    for (int it = 0; it < 32; ++it) {
        const int kb  = it << 5;
        const int cur = it & 1;
        if (wave == 0)
            __builtin_amdgcn_s_wait_tensorcnt(0);
        __syncthreads();
        if (wave == 0 && it + 1 < 32)
            tdm_load_2d((unsigned int)(size_t)&Bst[1 - cur][0],
                        wbase + kb + 32, 32, 128, DD);

        Frag a;
        load_a_bf16(a, ar + kb + half * 16, ar + kb + half * 16 + 8);
        const unsigned short* bs = &Bst[cur][0] + l16 * 32 + half * 16;
#pragma unroll
        for (int j = 0; j < 8; ++j) {
            Frag b;
            b.u4[0] = *(const u32x4*)(bs + j * 512);
            b.u4[1] = *(const u32x4*)(bs + j * 512 + 8);
            acc[j] = wmma_bf16(a, b, acc[j]);
        }
        __syncthreads();
    }

#pragma unroll
    for (int j = 0; j < 8; ++j) {
        const int gn = tn + j * 16 + l16;
        const float bias = out_b[gn];
#pragma unroll
        for (int v = 0; v < 8; ++v) {
            const int gm = tm + v + half * 8;
            out[(size_t)gm * DD + gn] = acc[j][v] + bias;
        }
    }
}

// ---------------------------------------------------------------------------
// Launch.  Workspace (30 MB):
//   [0,4M) kvwT | [4,6M) owT | [6,14M) Kh | [14,22M) Vt | [22,30M) attn
// ---------------------------------------------------------------------------
extern "C" void kernel_launch(void* const* d_in, const int* in_sizes, int n_in,
                              void* d_out, int out_size, void* d_ws, size_t ws_size,
                              hipStream_t stream) {
    const float* x     = (const float*)d_in[0];
    const float* q     = (const float*)d_in[1];
    const float* kv_w  = (const float*)d_in[2];
    const float* kv_b  = (const float*)d_in[3];
    const float* out_w = (const float*)d_in[4];
    const float* out_b = (const float*)d_in[5];
    float* out = (float*)d_out;

    char* ws = (char*)d_ws;
    unsigned short* kvwT = (unsigned short*)(ws);
    unsigned short* owT  = (unsigned short*)(ws + (size_t)(4u  << 20));
    unsigned short* Kh   = (unsigned short*)(ws + (size_t)(6u  << 20));
    unsigned short* Vt   = (unsigned short*)(ws + (size_t)(14u << 20));
    unsigned short* attn = (unsigned short*)(ws + (size_t)(22u << 20));

    prep_kernel<<<2048, 256, 0, stream>>>(kv_w, out_w, kvwT, owT);
    kv_proj_kernel<<<512, 256, 0, stream>>>(x, kv_b, kvwT, Kh, Vt);    // 512 block-tiles
    attn_kernel<<<512, 256, 0, stream>>>(q, Kh, Vt, attn);             // 4096 wave-tiles
    out_proj_kernel<<<256, 256, 0, stream>>>(attn, owT, out_b, out);   // 256 block-tiles
}
